// embedding_network_47107201303308
// MI455X (gfx1250) — matrix operand order repaired and checked by reading, not compiled, not run on hardware
//
#include <hip/hip_runtime.h>

typedef __bf16 bf16_t;
typedef bf16_t bfx16 __attribute__((ext_vector_type(16)));
typedef bf16_t bfx8  __attribute__((ext_vector_type(8)));
typedef float  f32x8 __attribute__((ext_vector_type(8)));

#define NV 1024
#define BATCH 32
#define EMBD 64

__device__ __forceinline__ f32x8 wmma_bf16(bfx16 a, bfx16 b, f32x8 c) {
  return __builtin_amdgcn_wmma_f32_16x16x32_bf16(false, a, false, b, (short)0, c, false, false);
}
__device__ __forceinline__ bfx16 join8(bfx8 lo, bfx8 hi) {
  return __builtin_shufflevector(lo, hi, 0,1,2,3,4,5,6,7,8,9,10,11,12,13,14,15);
}

// ---- 1) graph f32 -> bf16 (64 MB, lives in L2 for the 3 WMMA passes) ----
__global__ void __launch_bounds__(256)
conv_graph_kernel(const float* __restrict__ g, bf16_t* __restrict__ gb) {
  size_t base = ((size_t)blockIdx.x * 256 + threadIdx.x) * 8;
  const float4* p = (const float4*)(g + base);
  float4 a = p[0], b = p[1];
  bfx8 o;
  o[0]=(bf16_t)a.x; o[1]=(bf16_t)a.y; o[2]=(bf16_t)a.z; o[3]=(bf16_t)a.w;
  o[4]=(bf16_t)b.x; o[5]=(bf16_t)b.y; o[6]=(bf16_t)b.z; o[7]=(bf16_t)b.w;
  *(bfx8*)(gb + base) = o;
}

// ---- 2) W2 -> bf16 row-major; c34[e] = sum_k W3[e,k]*W4[k] ----
__global__ void __launch_bounds__(256)
prep_small_kernel(const float* __restrict__ W2, const float* __restrict__ W3,
                  const float* __restrict__ W4, bf16_t* __restrict__ W2b,
                  float* __restrict__ c34) {
  int t = threadIdx.x;
  for (int k = t; k < EMBD*EMBD; k += 256) W2b[k] = (bf16_t)W2[k];
  if (t < EMBD) {
    float s = 0.f;
    #pragma unroll
    for (int k = 0; k < EMBD; ++k) s += W3[t*EMBD + k] * W4[k];
    c34[t] = s;
  }
}

// ---- 3) deg row-sum + baseT[b,e,i] (f32) + embT0 = relu(base) (bf16) ----
__global__ void __launch_bounds__(256)
base_kernel(const float* __restrict__ g, const float* __restrict__ Xv,
            const float* __restrict__ W1, const float* __restrict__ c34,
            float* __restrict__ baseT, bf16_t* __restrict__ embT0) {
  int wave = threadIdx.x >> 5, lane = threadIdx.x & 31;
  int row = blockIdx.x * 8 + wave;          // b*1024 + i
  int b = row >> 10, i = row & (NV - 1);
  const float* gr = g + (size_t)row * NV;
  float s = 0.f;
  for (int j = lane; j < NV; j += 32) s += gr[j];
  #pragma unroll
  for (int m = 16; m > 0; m >>= 1) s += __shfl_xor(s, m, 32);
  float xv = Xv[row];
  #pragma unroll
  for (int t = 0; t < 2; ++t) {
    int e = lane + t * 32;
    float v = xv * W1[e] + s * c34[e];
    size_t off = ((size_t)b * EMBD + e) * NV + i;
    baseT[off]  = v;
    embT0[off]  = (bf16_t)(v > 0.f ? v : 0.f);
  }
}

// ---- 4) one message-passing step, all-WMMA, software-pipelined ----
// nsT[e,i] = sum_j embT[e,j] * G[i,j]   (A = embT rows, B = graph rows -> all contiguous)
// embT_out = relu(baseT + W2 @ nsT)     (nsT staged via LDS in B-layout)
#define LOADB(bt, j0)                                                         \
  bt = join8(*(const bfx8*)(gRow + (j0) + hf*8),                              \
             *(const bfx8*)(gRow + (j0) + hf*8 + 16));
#define LOADA(d0, d1, d2, d3, j0) {                                           \
  const bf16_t* ab = eT + (j0) + hf*8;                                        \
  const bf16_t* p0 = ab + (size_t)(r     ) * NV;                              \
  const bf16_t* p1 = ab + (size_t)(r + 16) * NV;                              \
  const bf16_t* p2 = ab + (size_t)(r + 32) * NV;                              \
  const bf16_t* p3 = ab + (size_t)(r + 48) * NV;                              \
  d0 = join8(*(const bfx8*)p0, *(const bfx8*)(p0 + 16));                      \
  d1 = join8(*(const bfx8*)p1, *(const bfx8*)(p1 + 16));                      \
  d2 = join8(*(const bfx8*)p2, *(const bfx8*)(p2 + 16));                      \
  d3 = join8(*(const bfx8*)p3, *(const bfx8*)(p3 + 16)); }

__global__ void __launch_bounds__(256)
step_kernel(const bf16_t* __restrict__ gB, const bf16_t* __restrict__ embT,
            const float* __restrict__ baseT, const bf16_t* __restrict__ W2b,
            bf16_t* __restrict__ embT_out) {
  __shared__ bf16_t lds_ns[8][16][EMBD];   // [wave][i_local][f], 16 KB
  const int tid  = threadIdx.x;
  const int wave = tid >> 5;
  const int lane = tid & 31;
  const int r    = lane & 15;
  const int hf   = lane >> 4;
  const int b    = blockIdx.x >> 3;
  const int i0   = (blockIdx.x & 7) * 128 + wave * 16;

  const bf16_t* gRow = gB + ((size_t)b * NV + (i0 + r)) * NV; // B-tile: lane holds G[i0+r, j..]
  const bf16_t* eT   = embT + (size_t)b * EMBD * NV;

  f32x8 acc0 = {}, acc1 = {}, acc2 = {}, acc3 = {};
  bfx16 btA, A0, A1, A2, A3;
  bfx16 btB, B0, B1, B2, B3;

  // prologue: chunk 0 into set A
  LOADB(btA, 0);
  LOADA(A0, A1, A2, A3, 0);

  for (int j0 = 0; j0 < NV; j0 += 64) {
    // prefetch chunk j0+32 into set B, then compute on set A
    LOADB(btB, j0 + 32);
    LOADA(B0, B1, B2, B3, j0 + 32);
    acc0 = wmma_bf16(A0, btA, acc0);
    acc1 = wmma_bf16(A1, btA, acc1);
    acc2 = wmma_bf16(A2, btA, acc2);
    acc3 = wmma_bf16(A3, btA, acc3);
    // prefetch chunk j0+64 into set A (final iter over-reads 32 elems into
    // adjacent workspace buffer -- allocated, values unused), compute set B
    LOADB(btA, j0 + 64);
    LOADA(A0, A1, A2, A3, j0 + 64);
    acc0 = wmma_bf16(B0, btB, acc0);
    acc1 = wmma_bf16(B1, btB, acc1);
    acc2 = wmma_bf16(B2, btB, acc2);
    acc3 = wmma_bf16(B3, btB, acc3);
  }

  // stage nsT tiles as bf16, i-major, per-wave private region
  #pragma unroll
  for (int v = 0; v < 8; ++v) {
    lds_ns[wave][r][ 0 + hf*8 + v] = (bf16_t)acc0[v];
    lds_ns[wave][r][16 + hf*8 + v] = (bf16_t)acc1[v];
    lds_ns[wave][r][32 + hf*8 + v] = (bf16_t)acc2[v];
    lds_ns[wave][r][48 + hf*8 + v] = (bf16_t)acc3[v];
  }
  __syncthreads();

  // reload nsT in WMMA B-layout (table order along f per lane)
  const bf16_t* pns = &lds_ns[wave][r][hf * 8];
  bfx16 nsB0 = join8(*(const bfx8*)(pns     ), *(const bfx8*)(pns + 16));
  bfx16 nsB1 = join8(*(const bfx8*)(pns + 32), *(const bfx8*)(pns + 48));

  #pragma unroll
  for (int ett = 0; ett < 4; ++ett) {
    f32x8 acc = {};
    const bf16_t* w2r = W2b + (ett*16 + r) * EMBD + hf*8;   // A = W2 rows
    acc = wmma_bf16(join8(*(const bfx8*)(w2r     ), *(const bfx8*)(w2r + 16)), nsB0, acc);
    acc = wmma_bf16(join8(*(const bfx8*)(w2r + 32), *(const bfx8*)(w2r + 48)), nsB1, acc);
    const size_t ob = ((size_t)b*EMBD + ett*16 + hf*8) * NV + i0 + r;
    const float* bp = baseT + ob;
    bf16_t* op = embT_out + ob;
    #pragma unroll
    for (int v = 0; v < 8; ++v) {
      float x = acc[v] + bp[(size_t)v * NV];
      op[(size_t)v * NV] = (bf16_t)(x > 0.f ? x : 0.f);
    }
  }
}

// ---- 5) per-batch: emb_sum -> s6[b] = dot(W5[:64], relu(W6 @ emb_sum)) ----
__global__ void __launch_bounds__(64)
reduce6_kernel(const bf16_t* __restrict__ embT, const float* __restrict__ W6,
               const float* __restrict__ W5, float* __restrict__ s6) {
  __shared__ float es[EMBD];
  __shared__ float hs[EMBD];
  int b = blockIdx.x, e = threadIdx.x;
  const bf16_t* p = embT + ((size_t)b * EMBD + e) * NV;
  float s = 0.f;
  for (int i = 0; i < NV; ++i) s += (float)p[i];
  es[e] = s;
  __syncthreads();
  float h = 0.f;
  #pragma unroll
  for (int k = 0; k < EMBD; ++k) h += W6[e*EMBD + k] * es[k];
  hs[e] = W5[e] * (h > 0.f ? h : 0.f);
  __syncthreads();
  if (e == 0) {
    float t = 0.f;
    #pragma unroll
    for (int k = 0; k < EMBD; ++k) t += hs[k];
    s6[b] = t;
  }
}

// ---- 6) out[b,i] = s6[b] + dot(W5[64:], relu(W7 @ embT[:,i])) ----
__global__ void __launch_bounds__(256)
out_kernel(const bf16_t* __restrict__ embT, const float* __restrict__ W7,
           const float* __restrict__ W5, const float* __restrict__ s6,
           float* __restrict__ out) {
  int idx = blockIdx.x * 256 + threadIdx.x;   // b*1024 + i
  int b = idx >> 10;
  const bf16_t* p = embT + (size_t)b * EMBD * NV + (idx & (NV - 1));
  float t[EMBD];
  #pragma unroll
  for (int e = 0; e < EMBD; ++e) t[e] = (float)p[(size_t)e * NV];
  float acc = s6[b];
  for (int ep = 0; ep < EMBD; ++ep) {
    const float* wr = W7 + ep * EMBD;
    float h = 0.f;
    #pragma unroll
    for (int k = 0; k < EMBD; ++k) h += wr[k] * t[k];
    acc += W5[EMBD + ep] * (h > 0.f ? h : 0.f);
  }
  out[idx] = acc;
}

extern "C" void kernel_launch(void* const* d_in, const int* in_sizes, int n_in,
                              void* d_out, int out_size, void* d_ws, size_t ws_size,
                              hipStream_t stream) {
  const float* graph = (const float*)d_in[0];
  const float* Xv    = (const float*)d_in[1];
  const float* W1    = (const float*)d_in[2];
  const float* W2    = (const float*)d_in[3];
  const float* W3    = (const float*)d_in[4];
  const float* W4    = (const float*)d_in[5];
  const float* W5    = (const float*)d_in[6];
  const float* W6    = (const float*)d_in[7];
  const float* W7    = (const float*)d_in[8];
  (void)in_sizes; (void)n_in; (void)out_size; (void)ws_size;

  char* ws = (char*)d_ws;
  bf16_t* gBF   = (bf16_t*)(ws);                 // 64 MB
  bf16_t* embA  = (bf16_t*)(ws + 67108864);      // 4 MB  [b,e,i] bf16
  bf16_t* embB  = (bf16_t*)(ws + 71303168);      // 4 MB
  float*  baseT = (float*) (ws + 75497472);      // 8 MB  [b,e,i] f32
  bf16_t* W2b   = (bf16_t*)(ws + 83886080);      // 8 KB
  float*  c34   = (float*) (ws + 83894272);      // 256 B
  float*  s6    = (float*) (ws + 83894528);      // 128 B

  conv_graph_kernel<<<16384, 256, 0, stream>>>(graph, gBF);
  prep_small_kernel<<<1, 256, 0, stream>>>(W2, W3, W4, W2b, c34);
  base_kernel<<<4096, 256, 0, stream>>>(graph, Xv, W1, c34, baseT, embA);
  // step 1 folded into base_kernel (ns==0). Steps 2..4:
  step_kernel<<<256, 256, 0, stream>>>(gBF, embA, baseT, W2b, embB);
  step_kernel<<<256, 256, 0, stream>>>(gBF, embB, baseT, W2b, embA);
  step_kernel<<<256, 256, 0, stream>>>(gBF, embA, baseT, W2b, embB);
  reduce6_kernel<<<32, 64, 0, stream>>>(embB, W6, W5, s6);
  out_kernel<<<128, 256, 0, stream>>>(embB, W7, W5, s6, (float*)d_out);
}
